// spaMultiAttn_19018115186866
// MI455X (gfx1250) — matrix operand order repaired and checked by reading, not compiled, not run on hardware
//
#include <hip/hip_runtime.h>

typedef __attribute__((ext_vector_type(16))) __bf16 v16bf;
typedef __attribute__((ext_vector_type(8)))  __bf16 v8bf;
typedef __attribute__((ext_vector_type(8)))  float  v8f;

#define LA_EPS 1e-7f
#define BROW 392   // padded LDS row stride in bf16 elements (784 B = 49*16B)

// ---------------- fp32 -> bf16 convert ----------------
__global__ void cvt_bf16_kernel(const float* __restrict__ in,
                                __bf16* __restrict__ out, int n) {
  int i = blockIdx.x * 256 + threadIdx.x;
  if (i < n) out[i] = (__bf16)in[i];
}

// ---------------- WMMA GEMM: C[M,N] = A[M,K] * Bt[N,K]^T + bias[N] ----------
// A, Bt row-major bf16; C row-major fp32. Block = 256 threads (8 waves).
// Block tile = 256(M) x 64(N); wave tile = 32(M) x 64(N) -> 8 accumulators.
// B panel (64 x K) staged once in LDS (padded rows, conflict-free fragment
// reads). Both the A fragments (global) and the B fragments (LDS) are
// double-buffered one k-step ahead so the 8-WMMA block never waits on a
// just-issued load. K is compile time -> full unroll, constant offsets.
// grid = (M/256, N/64).
template <int K>
__global__ __launch_bounds__(256)
void gemm_bf16_kernel(const __bf16* __restrict__ A,
                      const __bf16* __restrict__ Bt,
                      const float*  __restrict__ bias,
                      float*        __restrict__ C,
                      int N) {
  __shared__ __bf16 Bs[64 * BROW];

  const int tid  = threadIdx.x;
  const int lane = tid & 31;
  const int wv   = tid >> 5;
  const int l16  = lane & 15;
  const int lhi  = lane >> 4;
  const int nb   = blockIdx.y * 64;
  const int mrow = blockIdx.x * 256 + wv * 32;

  // ---- cooperative B panel load: 64 rows x K bf16, 16B chunks ----
  {
    constexpr int CH = K >> 3;                    // 16B chunks per row
    for (int e = tid; e < 64 * CH; e += 256) {
      int r = e / CH, c = e % CH;
      *(v8bf*)(Bs + r * BROW + c * 8) =
          *(const v8bf*)(Bt + (size_t)(nb + r) * K + c * 8);
    }
  }

  // A fragment feed (ISA A layout): lane row = m+l16, k offsets {lhi*8, lhi*8+16}
  const __bf16* Arow0 = A + (size_t)(mrow + l16) * K + lhi * 8;
  const __bf16* Arow1 = Arow0 + (size_t)16 * K;

  v8f acc[2][4];
#pragma unroll
  for (int t = 0; t < 4; ++t) {
    float bv = bias[nb + t * 16 + l16];
#pragma unroll
    for (int r = 0; r < 8; ++r) { acc[0][t][r] = bv; acc[1][t][r] = bv; }
  }

  // preload first A fragments (global)
  v8bf a0lo = *(const v8bf*)(Arow0);
  v8bf a0hi = *(const v8bf*)(Arow0 + 16);
  v8bf a1lo = *(const v8bf*)(Arow1);
  v8bf a1hi = *(const v8bf*)(Arow1 + 16);

  __syncthreads();

  // preload first B fragments (LDS)
  const __bf16* Bbase = Bs + (size_t)l16 * BROW + lhi * 8;
  v16bf bcur[4];
#pragma unroll
  for (int t = 0; t < 4; ++t) {
    v8bf b0 = *(const v8bf*)(Bbase + t * 16 * BROW);
    v8bf b1 = *(const v8bf*)(Bbase + t * 16 * BROW + 16);
#pragma unroll
    for (int i = 0; i < 8; ++i) { bcur[t][i] = b0[i]; bcur[t][i + 8] = b1[i]; }
  }

#pragma unroll
  for (int k0 = 0; k0 < K; k0 += 32) {
    const int kn = (k0 + 32 < K) ? (k0 + 32) : 0;  // constant-folded

    // prefetch next A fragments (global)
    v8bf p0lo = *(const v8bf*)(Arow0 + kn);
    v8bf p0hi = *(const v8bf*)(Arow0 + kn + 16);
    v8bf p1lo = *(const v8bf*)(Arow1 + kn);
    v8bf p1hi = *(const v8bf*)(Arow1 + kn + 16);

    // prefetch next B fragments (LDS)
    v16bf bnxt[4];
#pragma unroll
    for (int t = 0; t < 4; ++t) {
      v8bf b0 = *(const v8bf*)(Bbase + kn + t * 16 * BROW);
      v8bf b1 = *(const v8bf*)(Bbase + kn + t * 16 * BROW + 16);
#pragma unroll
      for (int i = 0; i < 8; ++i) { bnxt[t][i] = b0[i]; bnxt[t][i + 8] = b1[i]; }
    }

    v16bf a0, a1;
#pragma unroll
    for (int i = 0; i < 8; ++i) {
      a0[i] = a0lo[i]; a0[i + 8] = a0hi[i];
      a1[i] = a1lo[i]; a1[i + 8] = a1hi[i];
    }

#pragma unroll
    for (int t = 0; t < 4; ++t) {
      acc[0][t] = __builtin_amdgcn_wmma_f32_16x16x32_bf16(
          false, a0, false, bcur[t], (short)0, acc[0][t], false, false);
      acc[1][t] = __builtin_amdgcn_wmma_f32_16x16x32_bf16(
          false, a1, false, bcur[t], (short)0, acc[1][t], false, false);
    }

    a0lo = p0lo; a0hi = p0hi; a1lo = p1lo; a1hi = p1hi;
#pragma unroll
    for (int t = 0; t < 4; ++t) bcur[t] = bnxt[t];
  }

  // C/D layout: element r of lane L -> M = r + 8*(L>>4), N = L&15
#pragma unroll
  for (int s = 0; s < 2; ++s) {
    float* Crow = C + (size_t)(mrow + s * 16 + lhi * 8) * N + nb + l16;
#pragma unroll
    for (int t = 0; t < 4; ++t)
#pragma unroll
      for (int r = 0; r < 8; ++r)
        Crow[(size_t)r * N + t * 16] = acc[s][t][r];
  }
}

// ---------------- column softmax over token axis (n=4096) -------------------
// qkv layout: [b*4096, 1152]; q cols [0,384), k cols [384,768).
// grid = 256 blocks: (tsel in {q,k}) x (b*8+h). block = 288 = (48 ch) x (6 slices)
__global__ __launch_bounds__(288)
void softmax_cols_kernel(float* __restrict__ qkv) {
  const int LD = 1152, NTOK = 4096;
  int tsel = blockIdx.x >> 7;          // 0 = q, 1 = k
  int bh   = blockIdx.x & 127;
  int b = bh >> 3, h = bh & 7;
  int j   = threadIdx.x % 48;
  int sub = threadIdx.x / 48;          // 0..5

  float* base = qkv + (size_t)b * NTOK * LD + tsel * 384 + h * 48 + j;
  __shared__ float red[6][48];

  float m = -3.4e38f;
  for (int n = sub; n < NTOK; n += 6) m = fmaxf(m, base[(size_t)n * LD]);
  red[sub][j] = m;
  __syncthreads();
  float mm = red[0][j];
#pragma unroll
  for (int s = 1; s < 6; ++s) mm = fmaxf(mm, red[s][j]);
  __syncthreads();

  float sum = 0.f;
  for (int n = sub; n < NTOK; n += 6) {
    float e = expf(base[(size_t)n * LD] - mm);
    base[(size_t)n * LD] = e;
    sum += e;
  }
  red[sub][j] = sum;
  __syncthreads();
  float ss = 0.f;
#pragma unroll
  for (int s = 0; s < 6; ++s) ss += red[s][j];
  float inv = 1.f / ss;
  for (int n = sub; n < NTOK; n += 6) base[(size_t)n * LD] *= inv;
}

// ---------------- kv[b,h] = k_sm^T @ v  (48x48, K=4096 reduction) -----------
// grid = 128 (b*8+h), block = 256; each thread owns 9 of the 2304 cells.
__global__ __launch_bounds__(256)
void kv_kernel(const float* __restrict__ qkv, float* __restrict__ kv) {
  const int LD = 1152;
  int bh = blockIdx.x;
  int b = bh >> 3, h = bh & 7;
  const float* kbase = qkv + (size_t)b * 4096 * LD + 384 + h * 48;
  const float* vbase = qkv + (size_t)b * 4096 * LD + 768 + h * 48;

  __shared__ float kt[32][48];
  __shared__ float vt[32][48];
  float acc[9];
#pragma unroll
  for (int q = 0; q < 9; ++q) acc[q] = 0.f;
  int tid = threadIdx.x;

  for (int n0 = 0; n0 < 4096; n0 += 32) {
    for (int e = tid; e < 32 * 48; e += 256) {
      int r = e / 48, c = e % 48;
      kt[r][c] = kbase[(size_t)(n0 + r) * LD + c];
      vt[r][c] = vbase[(size_t)(n0 + r) * LD + c];
    }
    __syncthreads();
#pragma unroll
    for (int q = 0; q < 9; ++q) {
      int cell = tid * 9 + q;
      int i = cell / 48, jj = cell % 48;
      float a = 0.f;
#pragma unroll 8
      for (int r = 0; r < 32; ++r) a += kt[r][i] * vt[r][jj];
      acc[q] += a;
    }
    __syncthreads();
  }
#pragma unroll
  for (int q = 0; q < 9; ++q)
    kv[(size_t)bh * 2304 + tid * 9 + q] = acc[q];
}

// ---------------- attn = (q_sm @ kv) / (rowsum(q_sm)+eps), merge heads ------
// Writes attn bf16 [b*4096, 384]. grid = (128 bh, 8 chunks of 512 tokens),
// block = 288 = (48 out-ch) x (6 tokens).
__global__ __launch_bounds__(288)
void attn_kernel(const float* __restrict__ qkv, const float* __restrict__ kv,
                 __bf16* __restrict__ attn) {
  const int LD = 1152;
  int bh = blockIdx.x;
  int b = bh >> 3, h = bh & 7;
  int start = blockIdx.y * 512, end = start + 512;

  __shared__ float skv[48 * 48];
  __shared__ float qrow[6][48];
  int tid = threadIdx.x;
  for (int e = tid; e < 2304; e += 288) skv[e] = kv[(size_t)bh * 2304 + e];

  int j = tid % 48, sub = tid / 48;
  const float* qbase = qkv + (size_t)b * 4096 * LD + h * 48;
  __bf16* obase = attn + (size_t)b * 4096 * 384 + h * 48;
  __syncthreads();

  for (int nbase = start; nbase < end; nbase += 6) {
    int n = nbase + sub;
    bool ok = (n < end);
    qrow[sub][j] = ok ? qbase[(size_t)n * LD + j] : 0.f;
    __syncthreads();
    if (ok) {
      float denom = LA_EPS;
      float num = 0.f;
#pragma unroll 8
      for (int i = 0; i < 48; ++i) {
        float qi = qrow[sub][i];
        denom += qi;                  // == q @ colsum(k_softmax) (=1 per ch)
        num += qi * skv[i * 48 + j];
      }
      obase[(size_t)n * 384 + j] = (__bf16)(num / denom);
    }
    __syncthreads();
  }
}

// ---------------- launch ----------------------------------------------------
extern "C" void kernel_launch(void* const* d_in, const int* in_sizes, int n_in,
                              void* d_out, int out_size, void* d_ws, size_t ws_size,
                              hipStream_t stream) {
  const float* x      = (const float*)d_in[0];   // [16,64,64,384]
  const float* w_qkv  = (const float*)d_in[1];   // [1152,384]
  const float* b_qkv  = (const float*)d_in[2];   // [1152]
  const float* w_proj = (const float*)d_in[3];   // [384,384]
  const float* b_proj = (const float*)d_in[4];   // [384]
  float* out = (float*)d_out;                    // [16,64,64,384]

  const int D = 384, N1 = 1152;
  const int M = 16 * 4096;                       // 65536 tokens

  // Workspace carve-up (offsets in bytes, all 256B-aligned)
  char* ws = (char*)d_ws;
  __bf16* xb     = (__bf16*)(ws + 0);            // 50,331,648 B
  __bf16* wqkvb  = (__bf16*)(ws + 50331648);     //    884,736 B
  __bf16* wprojb = (__bf16*)(ws + 51216384);     //    294,912 B
  float*  qkv    = (float*) (ws + 51511296);     // 301,989,888 B
  float*  kvbuf  = (float*) (ws + 353501184);    //  1,179,648 B
  __bf16* attn   = (__bf16*)(ws + 354680832);    // 50,331,648 B

  int n;
  n = M * D;  cvt_bf16_kernel<<<(n + 255) / 256, 256, 0, stream>>>(x, xb, n);
  n = N1 * D; cvt_bf16_kernel<<<(n + 255) / 256, 256, 0, stream>>>(w_qkv, wqkvb, n);
  n = D * D;  cvt_bf16_kernel<<<(n + 255) / 256, 256, 0, stream>>>(w_proj, wprojb, n);

  // qkv = x @ w_qkv^T + b_qkv   [65536 x 1152]
  gemm_bf16_kernel<384><<<dim3(M / 256, N1 / 64), 256, 0, stream>>>(xb, wqkvb, b_qkv, qkv, N1);

  // softmax over tokens for q and k (in place)
  softmax_cols_kernel<<<256, 288, 0, stream>>>(qkv);

  // kv = k^T v per (b,h)
  kv_kernel<<<128, 256, 0, stream>>>(qkv, kvbuf);

  // attn = (q @ kv) / (q @ 1 + eps), heads merged, bf16
  attn_kernel<<<dim3(128, 8), 288, 0, stream>>>(qkv, kvbuf, attn);

  // out = attn @ w_proj^T + b_proj   [65536 x 384]
  gemm_bf16_kernel<384><<<dim3(M / 256, D / 64), 256, 0, stream>>>(attn, wprojb, b_proj, out, D);
}
